// DAttentionBaseline_39204461478726
// MI455X (gfx1250) — matrix operand order repaired and checked by reading, not compiled, not run on hardware
//
#include <hip/hip_runtime.h>
#include <hip/hip_bf16.h>

// ---------------------------------------------------------------------------
// Deformable attention forward for MI455X (gfx1250), wave32, WMMA bf16.
// B=8, C=512, L=2048, H=8, d=64, g=4, Lk=512, ngc=128, table=4095.
// Round 4: 6-arg tensor_load_to_lds (clang-23 signature); native bf16 cvts;
// async-to-LDS pos staging; bf16 LDS fragments = 2 x ds_load_b128, P = 4 x f32 b128 + cvt.
// ---------------------------------------------------------------------------

#define B_     8
#define C_     512
#define L_     2048
#define H_     8
#define D_     64
#define G_     4
#define NGC_   128
#define LK_    512
#define TBL_   4095

typedef __attribute__((ext_vector_type(16))) __bf16 v16bf;
typedef __attribute__((ext_vector_type(2)))  __bf16 v2bf;
typedef __attribute__((ext_vector_type(8)))  float  v8f;
typedef __attribute__((ext_vector_type(4)))  unsigned int u32x4;
typedef __attribute__((ext_vector_type(8)))  int i32x8;
typedef __attribute__((ext_vector_type(4)))  int i32x4;

union FragBF {
    unsigned short u[16];
    uint4 q[2];
    v16bf v;
};

// Native converts (v_cvt_bf16_f32 / v_cvt_pk_bf16_f32), RNE like the ref.
__device__ __forceinline__ unsigned short f2bf(float x) {
    __bf16 t = (__bf16)x;
    return __builtin_bit_cast(unsigned short, t);
}
__device__ __forceinline__ unsigned pack2(float a, float b) {
    v2bf t;
    t.x = (__bf16)a;
    t.y = (__bf16)b;
    return __builtin_bit_cast(unsigned, t);
}

__device__ __forceinline__ v8f wmma_bf16(const FragBF& a, const FragBF& b, v8f c) {
    return __builtin_amdgcn_wmma_f32_16x16x32_bf16(false, a.v, false, b.v,
                                                   (short)0, c, false, false);
}

// Low 32 bits of a generic pointer into LDS == wave-relative LDS byte offset.
__device__ __forceinline__ unsigned lds_off(const void* p) {
    return (unsigned)(size_t)p;
}

// ---------------------------------------------------------------------------
// Generic projection GEMM: Y[b,m,n] = sum_c W[m,c]*X[b,c,n] + bias[m]
// M = K = 512 fixed; N is 2048 (q/y) or 512 (k/v).
// Tiles: BM=128, BN=64, BK=32; 256 threads = 8 waves, 1 m-subtile per wave.
// As stored [m][k]; Bs stored TRANSPOSED [n][k] so B fragments are contiguous
// (every WMMA fragment = 2 x ds_load_b128).
// ---------------------------------------------------------------------------
__global__ __launch_bounds__(256) void dattn_gemm_kernel(
    const float* __restrict__ W, const float* __restrict__ X,
    const float* __restrict__ bias, float* __restrict__ Y, int N)
{
    __shared__ alignas(16) unsigned short As[128 * 32];   // [m][k]
    __shared__ alignas(16) unsigned short Bs[64 * 32];    // [n][k] (transposed)

    const int bm = blockIdx.y * 128;
    const int bn = blockIdx.x * 64;
    const int b  = blockIdx.z;
    const size_t xoff = (size_t)b * C_ * N;
    const int tid  = threadIdx.x;
    const int lane = tid & 31;
    const int w    = tid >> 5;
    const int half = lane >> 4;
    const int l16  = lane & 15;

    v8f acc[4] = {v8f{}, v8f{}, v8f{}, v8f{}};

    const int bn_n = tid & 63;          // B-stage: column n (coalesced lanes)
    const int bn_k = (tid >> 6) * 8;    // B-stage: k-octet

    for (int k0 = 0; k0 < 512; k0 += 32) {
        // ---- stage A tile (128x32) as bf16, float4 reads + uint2 LDS stores
        for (int i = 0; i < 4; ++i) {
            int idx4 = tid + i * 256;             // quad index
            int m  = idx4 >> 3;                   // 8 quads per row
            int kq = (idx4 & 7) * 4;
            float4 w4 = *(const float4*)&W[(size_t)(bm + m) * 512 + k0 + kq];
            uint2 p;
            p.x = pack2(w4.x, w4.y);
            p.y = pack2(w4.z, w4.w);
            *(uint2*)&As[m * 32 + kq] = p;
        }
        // ---- stage B tile transposed [n][k]: 8 coalesced-per-row global
        //      reads, one uint4 LDS store per thread
        {
            unsigned short t[8];
            for (int j = 0; j < 8; ++j)
                t[j] = f2bf(X[xoff + (size_t)(k0 + bn_k + j) * N + bn + bn_n]);
            uint4 p;
            p.x = (unsigned)t[0] | ((unsigned)t[1] << 16);
            p.y = (unsigned)t[2] | ((unsigned)t[3] << 16);
            p.z = (unsigned)t[4] | ((unsigned)t[5] << 16);
            p.w = (unsigned)t[6] | ((unsigned)t[7] << 16);
            *(uint4*)&Bs[bn_n * 32 + bn_k] = p;
        }
        if (k0 + 32 < 512)
            __builtin_prefetch(&X[xoff + (size_t)(k0 + 32 + bn_k) * N + bn + bn_n], 0, 1);
        __syncthreads();

        // A fragment: two contiguous 16B chunks -> 2 x ds_load_b128
        FragBF a;
        {
            const int row = (w * 16 + l16) * 32;
            a.q[0] = *(const uint4*)&As[row + half * 16];
            a.q[1] = *(const uint4*)&As[row + half * 16 + 8];
        }
        for (int nt = 0; nt < 4; ++nt) {
            FragBF bf;
            const int nrow = (nt * 16 + l16) * 32 + half * 16;
            bf.q[0] = *(const uint4*)&Bs[nrow];
            bf.q[1] = *(const uint4*)&Bs[nrow + 8];
            acc[nt] = wmma_bf16(a, bf, acc[nt]);
        }
        __syncthreads();
    }

    // C/D layout: VGPR r -> M = r + 8*half, N = l16
    for (int nt = 0; nt < 4; ++nt) {
        for (int r = 0; r < 8; ++r) {
            int m = bm + w * 16 + half * 8 + r;
            int n = bn + nt * 16 + l16;
            Y[(size_t)b * C_ * N + (size_t)m * N + n] = acc[nt][r] + bias[m];
        }
    }
}

// ---------------------------------------------------------------------------
// Offset network: depthwise conv (k=stride=4) -> LN(128ch) -> GELU -> 1x1 -> tanh
// One wave per (bg, n). Grid: (Lk/8, B*g), block 256 (8 waves).
// ---------------------------------------------------------------------------
__global__ __launch_bounds__(256) void dattn_offset_kernel(
    const float* __restrict__ q, const float* __restrict__ w_dw,
    const float* __restrict__ b_dw, const float* __restrict__ ln_g,
    const float* __restrict__ ln_b, const float* __restrict__ w_off,
    float* __restrict__ pos, float* __restrict__ pos_out,
    float* __restrict__ ref_out)
{
    const int w    = threadIdx.x >> 5;
    const int lane = threadIdx.x & 31;
    const int n    = blockIdx.x * 8 + w;
    const int bg   = blockIdx.y;

    float h[4];
    float s = 0.f, s2 = 0.f;
    for (int j = 0; j < 4; ++j) {
        int c = lane + j * 32;
        const float4 qv = *(const float4*)(q + ((size_t)(bg * NGC_ + c)) * L_ + n * 4);
        const float4 wv = *(const float4*)(w_dw + c * 4);
        float acc = b_dw[c] + qv.x * wv.x + qv.y * wv.y + qv.z * wv.z + qv.w * wv.w;
        h[j] = acc; s += acc; s2 += acc * acc;
    }
    for (int o = 16; o >= 1; o >>= 1) {
        s  += __shfl_xor(s, o);
        s2 += __shfl_xor(s2, o);
    }
    const float mu   = s * (1.0f / 128.0f);
    const float var  = s2 * (1.0f / 128.0f) - mu * mu;
    const float rstd = rsqrtf(var + 1e-5f);

    float dot = 0.f;
    for (int j = 0; j < 4; ++j) {
        int c = lane + j * 32;
        float hn = (h[j] - mu) * rstd * ln_g[c] + ln_b[c];
        float ge = 0.5f * hn * (1.0f + erff(hn * 0.70710678118654752f)); // exact GELU
        dot += ge * w_off[c];
    }
    for (int o = 16; o >= 1; o >>= 1) dot += __shfl_xor(dot, o);

    if (lane == 0) {
        float off = tanhf(dot) * (2.0f / (float)(LK_ - 1));
        float ref = (0.5f + (float)n) * (1.0f / (float)(LK_ - 1)) * 2.0f - 1.0f;
        float p = off + ref;
        pos[bg * LK_ + n]     = p;
        pos_out[bg * LK_ + n] = p;
        ref_out[bg * LK_ + n] = ref;
    }
}

// ---------------------------------------------------------------------------
// Deformed bilinear sampling of K/V source tokens (align_corners, zero pad).
// ---------------------------------------------------------------------------
__global__ __launch_bounds__(256) void dattn_gather_kernel(
    const float* __restrict__ x, const float* __restrict__ pos,
    float* __restrict__ xs)
{
    int idx = blockIdx.x * blockDim.x + threadIdx.x;
    if (idx >= B_ * C_ * LK_) return;
    int n = idx & (LK_ - 1);
    int c = (idx >> 9) & (C_ - 1);
    int b = idx >> 18;
    int bg = b * G_ + (c >> 7);

    float coord = pos[bg * LK_ + n];
    float ix  = (coord + 1.0f) * 0.5f * (float)(L_ - 1);
    float i0f = floorf(ix);
    float f   = ix - i0f;
    int i0 = (int)i0f, i1 = i0 + 1;
    const float* xr = x + (size_t)(b * C_ + c) * L_;
    float g0 = (i0 >= 0 && i0 < L_) ? xr[i0] : 0.0f;
    float g1 = (i1 >= 0 && i1 < L_) ? xr[i1] : 0.0f;
    xs[idx] = g0 * (1.0f - f) + g1 * f;
}

// ---------------------------------------------------------------------------
// Fused attention: per (b, h, 64-row tile of L): S = qT*K (WMMA), + RPE bias
// (bilinear from 4095-entry table), softmax over Lk=512, O = P*V^T (WMMA).
// LDS: S(64x512 f32) + ksT[n][d](bf16) + vs[d][n](bf16) + q tile + rpe + pos
// ~ 282 KB of the 320 KB/WGP. One WG/WGP, 8 wave32s.
// RPE row staged by the Tensor Data Mover (tensor_load_to_lds, TENSORcnt);
// pos staged by global_load_async_to_lds_b128 (ASYNCcnt).
// ---------------------------------------------------------------------------
__global__ __launch_bounds__(256) void dattn_attn_kernel(
    const float* __restrict__ q, const float* __restrict__ k,
    const float* __restrict__ v, const float* __restrict__ pos,
    const float* __restrict__ rpe, float* __restrict__ out)
{
    extern __shared__ char smem[];
    float* S  = (float*)smem;                          // 64*512 f32   = 128 KB
    float* rp = S + 64 * LK_;                          // 4096 f32     =  16 KB
    float* ps = rp + 4096;                             // 512 f32      =   2 KB
    unsigned short* ks = (unsigned short*)(ps + LK_);  // [n][d] bf16  =  64 KB
    unsigned short* vs = ks + 64 * LK_;                // [d][n] bf16  =  64 KB
    unsigned short* qs = vs + 64 * LK_;                // [m][d] bf16  =   8 KB

    const int m0 = blockIdx.x * 64;
    const int h  = blockIdx.y;
    const int b  = blockIdx.z;
    const int tid  = threadIdx.x;
    const int lane = tid & 31;
    const int w    = tid >> 5;
    const int half = lane >> 4;
    const int l16  = lane & 15;
    const int bg = b * G_ + (h >> 1);          // gh = H/g = 2
    const int c0 = b * C_ + h * D_;

    // ---- RPE row via Tensor Data Mover: 1D tile of 4095 f32 -> LDS (wave 0)
    if (w == 0) {
        unsigned long long ga = (unsigned long long)(size_t)(rpe + (size_t)h * TBL_);
        u32x4 g0;
        g0[0] = 1u;                                            // count=1, user D#
        g0[1] = lds_off(rp);                                   // lds_addr
        g0[2] = (unsigned)(ga & 0xFFFFFFFFu);                  // global_addr[31:0]
        g0[3] = (unsigned)((ga >> 32) & 0x1FFFFFFu)            // global_addr[56:32]
              | (2u << 30);                                    // type = 2 (image)
        i32x8 g1;
        g1[0] = (int)(2u << 16);                               // data_size = 4 bytes
        g1[1] = (int)((unsigned)(TBL_ & 0xFFFF) << 16);        // tensor_dim0 lo
        g1[2] = (int)(((unsigned)TBL_ >> 16) | (1u << 16));    // dim0 hi | tensor_dim1 lo = 1
        g1[3] = (int)((unsigned)TBL_ << 16);                   // dim1 hi=0 | tile_dim0 = 4095
        g1[4] = 1;                                             // tile_dim1 = 1
        g1[5] = TBL_;                                          // tensor_dim0_stride lo
        g1[6] = 0;
        g1[7] = 0;
        i32x4 z4 = {0, 0, 0, 0};
        i32x8 z8 = {0, 0, 0, 0, 0, 0, 0, 0};
        __builtin_amdgcn_tensor_load_to_lds(g0, g1, z4, z4, z8, 0);
        __builtin_amdgcn_s_wait_tensorcnt(0);
    }
    // ---- pos row via async global->LDS (128 lanes x b128)
    if (tid < 128) {
        unsigned off = lds_off(ps) + (unsigned)tid * 16u;
        const float* gp = pos + (size_t)bg * LK_ + tid * 4;
        asm volatile("global_load_async_to_lds_b128 %0, %1, off"
                     :: "v"(off), "v"(gp) : "memory");
    }
    asm volatile("s_wait_asynccnt 0x0" ::: "memory");

    // ---- stage K transposed [n][d]: coalesced global reads, uint4 LDS stores
    for (int i = 0; i < 16; ++i) {
        int o  = tid + i * 256;                // octet index (512 n x 8 octets)
        int n  = o & (LK_ - 1);
        int d8 = (o >> 9) * 8;
        unsigned short t[8];
        for (int j = 0; j < 8; ++j)
            t[j] = f2bf(k[(size_t)(c0 + d8 + j) * LK_ + n]);
        uint4 p;
        p.x = (unsigned)t[0] | ((unsigned)t[1] << 16);
        p.y = (unsigned)t[2] | ((unsigned)t[3] << 16);
        p.z = (unsigned)t[4] | ((unsigned)t[5] << 16);
        p.w = (unsigned)t[6] | ((unsigned)t[7] << 16);
        *(uint4*)&ks[n * 64 + d8] = p;
    }
    // ---- stage V [d][n]: float4 reads, uint2 LDS stores
    for (int i = 0; i < 32; ++i) {
        int qi = tid + i * 256;                // quad index (64 d x 128 quads)
        int d  = qi >> 7;
        int nq = (qi & 127) * 4;
        float4 v4 = *(const float4*)&v[(size_t)(c0 + d) * LK_ + nq];
        uint2 p;
        p.x = pack2(v4.x, v4.y);
        p.y = pack2(v4.z, v4.w);
        *(uint2*)&vs[d * LK_ + nq] = p;
    }
    // ---- stage q tile transposed [m][d] (small, scalar)
    for (int i = tid; i < 64 * 64; i += 256) {
        int m = i >> 6, d = i & 63;
        qs[m * 64 + d] = f2bf(q[(size_t)(c0 + d) * L_ + m0 + m]);
    }
    __syncthreads();

    // ---- Phase 1: S = qT * K.  wave w: m-subtile w/2, n-subtiles (w&1)*16..+15
    {
        const int msub = w >> 1;
        const int qrow = (msub * 16 + l16) * 64;
        FragBF a0, a1;
        a0.q[0] = *(const uint4*)&qs[qrow + half * 16];
        a0.q[1] = *(const uint4*)&qs[qrow + half * 16 + 8];
        a1.q[0] = *(const uint4*)&qs[qrow + 32 + half * 16];
        a1.q[1] = *(const uint4*)&qs[qrow + 32 + half * 16 + 8];
        for (int j = 0; j < 16; ++j) {
            int n = ((w & 1) * 16 + j) * 16 + l16;
            FragBF b0, b1;
            b0.q[0] = *(const uint4*)&ks[n * 64 + half * 16];
            b0.q[1] = *(const uint4*)&ks[n * 64 + half * 16 + 8];
            b1.q[0] = *(const uint4*)&ks[n * 64 + 32 + half * 16];
            b1.q[1] = *(const uint4*)&ks[n * 64 + 32 + half * 16 + 8];
            v8f acc = {};
            acc = wmma_bf16(a0, b0, acc);
            acc = wmma_bf16(a1, b1, acc);
            for (int r = 0; r < 8; ++r)
                S[(msub * 16 + half * 8 + r) * LK_ + n] = acc[r];
        }
    }
    __syncthreads();

    // ---- Phase 2: scale + RPE bias + softmax. 4 lanes/row, 128 cols each.
    {
        const int row = tid >> 2;
        const int s4  = tid & 3;
        float* Srow = S + row * LK_;
        const float qg = (float)(m0 + row) / (float)(L_ - 1) * 2.0f - 1.0f;
        const float scale = 0.125f;                  // 1/sqrt(64)
        float mx = -3.0e38f;
        for (int n4 = s4 * 32; n4 < s4 * 32 + 32; ++n4) {
            float4 sv = *(float4*)&Srow[n4 * 4];
            float4 pv = *(const float4*)&ps[n4 * 4];
            float vv[4] = {sv.x, sv.y, sv.z, sv.w};
            float pp[4] = {pv.x, pv.y, pv.z, pv.w};
            for (int e = 0; e < 4; ++e) {
                float disp = (qg - pp[e]) * 0.5f;
                float ix   = (disp + 1.0f) * 0.5f * (float)(TBL_ - 1);
                float i0f  = floorf(ix);
                float f    = ix - i0f;
                int i0 = (int)i0f, i1 = i0 + 1;
                float g0 = (i0 >= 0 && i0 < TBL_) ? rp[i0] : 0.0f;
                float g1 = (i1 >= 0 && i1 < TBL_) ? rp[i1] : 0.0f;
                vv[e] = vv[e] * scale + g0 * (1.0f - f) + g1 * f;
                mx = fmaxf(mx, vv[e]);
            }
            sv.x = vv[0]; sv.y = vv[1]; sv.z = vv[2]; sv.w = vv[3];
            *(float4*)&Srow[n4 * 4] = sv;
        }
        mx = fmaxf(mx, __shfl_xor(mx, 1));
        mx = fmaxf(mx, __shfl_xor(mx, 2));
        float sum = 0.f;
        for (int n4 = s4 * 32; n4 < s4 * 32 + 32; ++n4) {
            float4 sv = *(float4*)&Srow[n4 * 4];
            sv.x = __expf(sv.x - mx); sv.y = __expf(sv.y - mx);
            sv.z = __expf(sv.z - mx); sv.w = __expf(sv.w - mx);
            sum += sv.x + sv.y + sv.z + sv.w;
            *(float4*)&Srow[n4 * 4] = sv;
        }
        sum += __shfl_xor(sum, 1);
        sum += __shfl_xor(sum, 2);
        float inv = 1.0f / sum;
        for (int n4 = s4 * 32; n4 < s4 * 32 + 32; ++n4) {
            float4 sv = *(float4*)&Srow[n4 * 4];
            sv.x *= inv; sv.y *= inv; sv.z *= inv; sv.w *= inv;
            *(float4*)&Srow[n4 * 4] = sv;
        }
    }
    __syncthreads();

    // ---- Phase 3: O = P * V^T. wave w: m-subtile w/2, d-subtiles (w&1)*2+{0,1}
    {
        const int msub  = w >> 1;
        const int dbase = (w & 1) * 32;
        v8f acc[2] = {v8f{}, v8f{}};
        for (int k0 = 0; k0 < LK_; k0 += 32) {
            FragBF a;
            const float* Srow = S + (msub * 16 + l16) * LK_ + k0;
            float4 sa = *(const float4*)&Srow[half * 16];
            float4 sb = *(const float4*)&Srow[half * 16 + 4];
            float4 sc = *(const float4*)&Srow[half * 16 + 8];
            float4 sd = *(const float4*)&Srow[half * 16 + 12];
            a.q[0].x = pack2(sa.x, sa.y); a.q[0].y = pack2(sa.z, sa.w);
            a.q[0].z = pack2(sb.x, sb.y); a.q[0].w = pack2(sb.z, sb.w);
            a.q[1].x = pack2(sc.x, sc.y); a.q[1].y = pack2(sc.z, sc.w);
            a.q[1].z = pack2(sd.x, sd.y); a.q[1].w = pack2(sd.z, sd.w);
            for (int t = 0; t < 2; ++t) {
                int d = dbase + t * 16 + l16;
                FragBF bf;
                bf.q[0] = *(const uint4*)&vs[d * LK_ + k0 + half * 16];
                bf.q[1] = *(const uint4*)&vs[d * LK_ + k0 + half * 16 + 8];
                acc[t] = wmma_bf16(a, bf, acc[t]);
            }
        }
        for (int t = 0; t < 2; ++t) {
            int d = dbase + t * 16 + l16;
            for (int r = 0; r < 8; ++r) {
                int m = m0 + msub * 16 + half * 8 + r;
                out[(size_t)(c0 + d) * L_ + m] = acc[t][r];
            }
        }
    }
}

// ---------------------------------------------------------------------------
extern "C" void kernel_launch(void* const* d_in, const int* in_sizes, int n_in,
                              void* d_out, int out_size, void* d_ws, size_t ws_size,
                              hipStream_t stream)
{
    const float* x    = (const float*)d_in[0];
    const float* wq   = (const float*)d_in[1];
    const float* bq   = (const float*)d_in[2];
    const float* wk   = (const float*)d_in[3];
    const float* bk   = (const float*)d_in[4];
    const float* wv   = (const float*)d_in[5];
    const float* bv   = (const float*)d_in[6];
    const float* wo   = (const float*)d_in[7];
    const float* bo   = (const float*)d_in[8];
    const float* w_dw = (const float*)d_in[9];
    const float* b_dw = (const float*)d_in[10];
    const float* ln_g = (const float*)d_in[11];
    const float* ln_b = (const float*)d_in[12];
    const float* w_off= (const float*)d_in[13];
    const float* rpe  = (const float*)d_in[14];
    float* out = (float*)d_out;

    // workspace carve (bytes, 256-aligned)
    char* ws = (char*)d_ws;
    size_t o = 0;
    auto take = [&](size_t bytes) -> char* {
        char* p = ws + o;
        o += (bytes + 255) & ~(size_t)255;
        return p;
    };
    float* qbuf = (float*)take((size_t)B_ * C_ * L_  * 4);  // 33.5 MB
    float* posb = (float*)take((size_t)B_ * G_ * LK_ * 4);  //  64 KB
    float* xsb  = (float*)take((size_t)B_ * C_ * LK_ * 4);  // 8.4 MB
    float* kbuf = (float*)take((size_t)B_ * C_ * LK_ * 4);  // 8.4 MB
    float* vbuf = (float*)take((size_t)B_ * C_ * LK_ * 4);  // 8.4 MB
    float* obuf = (float*)take((size_t)B_ * C_ * L_  * 4);  // 33.5 MB

    const size_t y_elems = (size_t)B_ * C_ * L_;            // 8388608
    const size_t p_elems = (size_t)B_ * G_ * LK_;           // 16384
    float* pos_out = out + y_elems;
    float* ref_out = out + y_elems + p_elems;

    dim3 blk(256);

    // 1) q = wq @ x + bq
    dattn_gemm_kernel<<<dim3(L_ / 64, C_ / 128, B_), blk, 0, stream>>>(wq, x, bq, qbuf, L_);
    // 2) offset network -> pos (+ tuple outputs pos_out / ref_out)
    dattn_offset_kernel<<<dim3(LK_ / 8, B_ * G_), blk, 0, stream>>>(
        qbuf, w_dw, b_dw, ln_g, ln_b, w_off, posb, pos_out, ref_out);
    // 3) deformed sampling x_s
    dattn_gather_kernel<<<dim3((B_ * C_ * LK_) / 256), blk, 0, stream>>>(x, posb, xsb);
    // 4) k, v projections
    dattn_gemm_kernel<<<dim3(LK_ / 64, C_ / 128, B_), blk, 0, stream>>>(wk, xsb, bk, kbuf, LK_);
    dattn_gemm_kernel<<<dim3(LK_ / 64, C_ / 128, B_), blk, 0, stream>>>(wv, xsb, bv, vbuf, LK_);
    // 5) fused attention (big-LDS: S + K + V + q + rpe + pos ~ 282 KB / WGP)
    size_t smem = (size_t)64 * LK_ * 4 + 4096 * 4 + LK_ * 4
                + (size_t)64 * LK_ * 2 * 2 + (size_t)64 * 64 * 2;
    dattn_attn_kernel<<<dim3(L_ / 64, H_, B_), blk, smem, stream>>>(
        qbuf, kbuf, vbuf, posb, rpe, obuf);
    // 6) y = wo @ out + bo
    dattn_gemm_kernel<<<dim3(L_ / 64, C_ / 128, B_), blk, 0, stream>>>(wo, obuf, bo, out, L_);
}
